// QAConv_19370302505054
// MI455X (gfx1250) — matrix operand order repaired and checked by reading, not compiled, not run on hardware
//
#include <hip/hip_runtime.h>

// ---------------------------------------------------------------------------
// QAConv-style scoring fully fused for gfx1250 (MI455X):
//   per (g,p): S = Ker_p(192x512) * Gal_g(512x192) via v_wmma_f32_16x16x32_f16
//   (f16x3 split for ~fp32 accuracy), dual-axis max pool, BN -> FC -> BN ->
//   sigmoid, one scalar per block. 2048 blocks x 512 threads (16 wave32).
//   Software-pipelined: next K-tile fetched into registers under the WMMAs.
// ---------------------------------------------------------------------------

typedef __attribute__((ext_vector_type(16))) _Float16 v16h;
typedef __attribute__((ext_vector_type(8)))  float    v8f;

#define NG 32
#define NP 64
#define ND 512
#define HW 192          // h*w = 24*8
#define KB 32           // K per WMMA step
#define LDA 40          // padded LDS row stride in halves (80 B, bank friendly)

struct FragU { union { v16h v; uint4 q[2]; }; };

__device__ __forceinline__ v16h frag2(const _Float16* base, int off0, int off1) {
    FragU u;
    u.q[0] = *(const uint4*)(base + off0);   // ds_load_b128
    u.q[1] = *(const uint4*)(base + off1);   // ds_load_b128
    return u.v;
}

union H4 { _Float16 h[4]; unsigned long long u; };

union SMem {
    // GEMM tile staging: A (ker, [y][k]) and B^T (gal^T, [x][k]), hi/lo f16
    struct {
        _Float16 Ah[HW * LDA];
        _Float16 Al[HW * LDA];
        _Float16 Bh[HW * LDA];
        _Float16 Bl[HW * LDA];
    } t;                                   // 61440 B
    // epilogue scratch (tiles are dead by then)
    struct {
        float rowpart[HW][4];              // per-row max of each 48-col strip
        float colpart[HW][4];              // per-col max of each 48-row strip
        float red[512];                    // tree reduction
    } r;                                   // 8192 B
};

__global__ __launch_bounds__(512, 1)
void qaconv_wmma_kernel(const float* __restrict__ gal,   // [g][d][hw]
                        const float* __restrict__ prob,  // [p][d][hw]
                        const float* __restrict__ bnw, const float* __restrict__ bnb,
                        const float* __restrict__ bnm, const float* __restrict__ bnv,
                        const float* __restrict__ fcw, const float* __restrict__ fcb,
                        const float* __restrict__ lw,  const float* __restrict__ lb,
                        const float* __restrict__ lm,  const float* __restrict__ lv,
                        float* __restrict__ out)
{
    __shared__ __align__(16) SMem sm;

    const int p    = blockIdx.x;
    const int g    = blockIdx.y;
    const int tid  = threadIdx.x;
    const int lane = tid & 31;
    const int wave = tid >> 5;
    const int wm   = wave >> 2;      // 0..3 : 48-row strip
    const int wn   = wave & 3;       // 0..3 : 48-col strip
    const int ml   = lane & 15;
    const int h16  = lane >> 4;      // 0 | 1

    const float* __restrict__ Ag = prob + (size_t)p * (ND * HW);  // [d][y]
    const float* __restrict__ Bg = gal  + (size_t)g * (ND * HW);  // [d][x]

    const v8f vzero = {0.f, 0.f, 0.f, 0.f, 0.f, 0.f, 0.f, 0.f};
    v8f acc[3][3];
#pragma unroll
    for (int i = 0; i < 3; ++i)
#pragma unroll
        for (int j = 0; j < 3; ++j) acc[i][j] = vzero;

    // Staging work items: 1536 = 192 xy * 8 d-quads; thread t handles 3 items,
    // each item = 4 consecutive d for one xy (coalesced over xy across lanes).
    int itXY[3], itDQ[3];
#pragma unroll
    for (int it = 0; it < 3; ++it) {
        int j = tid + it * 512;
        itDQ[it] = j / HW;               // 0..7 (K-invariant, hoisted)
        itXY[it] = j - itDQ[it] * HW;    // 0..191
    }

    float aR[3][4], bR[3][4];            // software pipeline registers
#pragma unroll
    for (int it = 0; it < 3; ++it) {
        size_t base = (size_t)(itDQ[it] * 4) * HW + itXY[it];
#pragma unroll
        for (int q = 0; q < 4; ++q) {
            aR[it][q] = Ag[base + (size_t)q * HW];
            bR[it][q] = Bg[base + (size_t)q * HW];
        }
    }

    for (int k0 = 0; k0 < ND; k0 += KB) {
        __syncthreads();                 // previous compute done before LDS reuse
        // f16 hi/lo split of the pipelined registers, vectorized ds_store_b64.
#pragma unroll
        for (int it = 0; it < 3; ++it) {
            const int off = itXY[it] * LDA + itDQ[it] * 4;   // halves, 8B aligned
            H4 ah4, al4, bh4, bl4;
#pragma unroll
            for (int q = 0; q < 4; ++q) {
                float a = aR[it][q];
                _Float16 ha = (_Float16)a;                   // v_cvt_f16_f32 (RNE)
                ah4.h[q] = ha;
                al4.h[q] = (_Float16)(a - (float)ha);
                float b = bR[it][q];
                _Float16 hb = (_Float16)b;
                bh4.h[q] = hb;
                bl4.h[q] = (_Float16)(b - (float)hb);
            }
            *(unsigned long long*)(sm.t.Ah + off) = ah4.u;
            *(unsigned long long*)(sm.t.Al + off) = al4.u;
            *(unsigned long long*)(sm.t.Bh + off) = bh4.u;
            *(unsigned long long*)(sm.t.Bl + off) = bl4.u;
        }
        __syncthreads();

        // Fragments per ISA 7.12.2 16-bit layouts.
        v16h ah[3], al[3], bh[3], bl[3];
#pragma unroll
        for (int t = 0; t < 3; ++t) {
            int m     = wm * 48 + t * 16 + ml;
            int abase = m * LDA + h16 * 8;       // K runs {base..+7, base+16..+23}
            ah[t] = frag2(sm.t.Ah, abase, abase + 16);
            al[t] = frag2(sm.t.Al, abase, abase + 16);
            int n     = wn * 48 + t * 16 + ml;
            int bbase = n * LDA + h16 * 8;       // K runs {base..+7, base+16..+23}
            bh[t] = frag2(sm.t.Bh, bbase, bbase + 16);
            bl[t] = frag2(sm.t.Bl, bbase, bbase + 16);
        }

        // Fetch next K-tile into registers; consumed after the next barrier, so
        // the global latency hides under the 27 WMMAs below.
        if (k0 + KB < ND) {
#pragma unroll
            for (int it = 0; it < 3; ++it) {
                size_t base = (size_t)(k0 + KB + itDQ[it] * 4) * HW + itXY[it];
#pragma unroll
                for (int q = 0; q < 4; ++q) {
                    aR[it][q] = Ag[base + (size_t)q * HW];
                    bR[it][q] = Bg[base + (size_t)q * HW];
                }
            }
        }
        if (k0 + 2 * KB < ND) {          // warm L2 two steps ahead (global_prefetch_b8)
            __builtin_prefetch(Ag + (size_t)(k0 + 2 * KB) * HW + tid * 12, 0, 1);
            __builtin_prefetch(Bg + (size_t)(k0 + 2 * KB) * HW + tid * 12, 0, 1);
        }

        // f16x3: hi*hi + hi*lo + lo*hi (lo*lo ~2^-22, dropped)
#pragma unroll
        for (int i = 0; i < 3; ++i)
#pragma unroll
            for (int j = 0; j < 3; ++j) {
                acc[i][j] = __builtin_amdgcn_wmma_f32_16x16x32_f16(
                    false, ah[i], false, bh[j], (short)0, acc[i][j], false, false);
                acc[i][j] = __builtin_amdgcn_wmma_f32_16x16x32_f16(
                    false, ah[i], false, bl[j], (short)0, acc[i][j], false, false);
                acc[i][j] = __builtin_amdgcn_wmma_f32_16x16x32_f16(
                    false, al[i], false, bh[j], (short)0, acc[i][j], false, false);
            }
    }

    __syncthreads();   // tiles dead -> epilogue scratch

    // C/D layout: acc[t][u][r] is element (row = wm*48+t*16+r+h16*8, col = wn*48+u*16+ml).
    // Per-row max over this wave's 48 columns (reduce tiles, then xor-shuffle low 4 bits).
    float rowv[3][8];
#pragma unroll
    for (int t = 0; t < 3; ++t)
#pragma unroll
        for (int r = 0; r < 8; ++r)
            rowv[t][r] = fmaxf(fmaxf(acc[t][0][r], acc[t][1][r]), acc[t][2][r]);
#pragma unroll
    for (int t = 0; t < 3; ++t)
#pragma unroll
        for (int r = 0; r < 8; ++r) {
#pragma unroll
            for (int s = 1; s < 16; s <<= 1)
                rowv[t][r] = fmaxf(rowv[t][r], __shfl_xor(rowv[t][r], s, 32));
        }
    if (ml == 0) {
#pragma unroll
        for (int t = 0; t < 3; ++t)
#pragma unroll
            for (int r = 0; r < 8; ++r)
                sm.r.rowpart[wm * 48 + t * 16 + r + h16 * 8][wn] = rowv[t][r];
    }

    // Per-col max over this wave's 48 rows (lane-local over tiles, then xor16 across halves).
    float colv[3];
#pragma unroll
    for (int t = 0; t < 3; ++t) {
        float m = acc[0][t][0];
#pragma unroll
        for (int i = 0; i < 3; ++i)
#pragma unroll
            for (int r = 0; r < 8; ++r)
                m = fmaxf(m, acc[i][t][r]);
        colv[t] = fmaxf(m, __shfl_xor(m, 16, 32));
    }
    if (h16 == 0) {
#pragma unroll
        for (int t = 0; t < 3; ++t)
            sm.r.colpart[wn * 48 + t * 16 + ml][wm] = colv[t];
    }
    __syncthreads();

    // score[i] layout: [0..HW) = max over y (col max), [HW..2HW) = max over x (row max).
    // Fold BN affine into FC dot; deterministic tree reduction.
    float contrib = 0.f;
    if (tid < 2 * HW) {
        float m;
        if (tid < HW) {
            m = fmaxf(fmaxf(sm.r.colpart[tid][0], sm.r.colpart[tid][1]),
                      fmaxf(sm.r.colpart[tid][2], sm.r.colpart[tid][3]));
        } else {
            int y = tid - HW;
            m = fmaxf(fmaxf(sm.r.rowpart[y][0], sm.r.rowpart[y][1]),
                      fmaxf(sm.r.rowpart[y][2], sm.r.rowpart[y][3]));
        }
        float bscale = bnw[0] * rsqrtf(bnv[0] + 1e-5f);
        contrib = ((m - bnm[0]) * bscale + bnb[0]) * fcw[tid];
    }
    sm.r.red[tid] = contrib;
    __syncthreads();
    for (int s = 256; s > 0; s >>= 1) {
        if (tid < s) sm.r.red[tid] += sm.r.red[tid + s];
        __syncthreads();
    }
    if (tid == 0) {
        float score  = sm.r.red[0] + fcb[0];
        float lscale = lw[0] * rsqrtf(lv[0] + 1e-5f);
        score = (score - lm[0]) * lscale + lb[0];
        out[(size_t)g * NP + p] = 1.0f / (1.0f + __expf(-score * 0.1f));
    }
}

extern "C" void kernel_launch(void* const* d_in, const int* in_sizes, int n_in,
                              void* d_out, int out_size, void* d_ws, size_t ws_size,
                              hipStream_t stream) {
    const float* gal  = (const float*)d_in[0];
    const float* prob = (const float*)d_in[1];
    const float* bnw  = (const float*)d_in[2];
    const float* bnb  = (const float*)d_in[3];
    const float* bnm  = (const float*)d_in[4];
    const float* bnv  = (const float*)d_in[5];
    const float* fcw  = (const float*)d_in[6];
    const float* fcb  = (const float*)d_in[7];
    const float* lw   = (const float*)d_in[8];
    const float* lb   = (const float*)d_in[9];
    const float* lm   = (const float*)d_in[10];
    const float* lv   = (const float*)d_in[11];

    dim3 grid(NP, NG);   // one block per (probe, gallery) pair
    qaconv_wmma_kernel<<<grid, 512, 0, stream>>>(
        gal, prob, bnw, bnb, bnm, bnv, fcw, fcb, lw, lb, lm, lv, (float*)d_out);
}
